// FujiAttention_1924145348774
// MI455X (gfx1250) — matrix operand order repaired and checked by reading, compiled, not run on hardware
//
#include <hip/hip_runtime.h>
#include <cstdint>
#include <cstddef>

typedef _Float16 half_t;
typedef _Float16 v16h __attribute__((ext_vector_type(16)));
typedef _Float16 v8h  __attribute__((ext_vector_type(8)));
typedef float    v8f  __attribute__((ext_vector_type(8)));
typedef int      v4i_t __attribute__((vector_size(16)));

#define D_HID   4096
#define NQ      32
#define NKV     8
#define HD      128
#define BATCH   4
#define SEQ     1024
#define TTOK    (BATCH*SEQ)
#define LN_THETA 13.122363377404328f     // ln(500000)
#define SM_SCALE 0.08838834764831845f    // 1/sqrt(128)

#if defined(__AMDGCN__) && __has_builtin(__builtin_amdgcn_global_load_async_to_lds_b128) && __has_builtin(__builtin_amdgcn_s_wait_asynccnt)
#define USE_ASYNC 1
#else
#define USE_ASYNC 0
#endif
#if defined(__AMDGCN__) && !USE_ASYNC
#warning "gfx1250 async-to-LDS builtins unavailable: falling back to VGPR staging"
#endif

__device__ __forceinline__ v16h cat16(v8h lo, v8h hi) {
  return __builtin_shufflevector(lo, hi, 0,1,2,3,4,5,6,7,8,9,10,11,12,13,14,15);
}
__device__ __forceinline__ v8f wmma_f16(v16h a, v16h b, v8f c) {
  return __builtin_amdgcn_wmma_f32_16x16x32_f16(false, a, false, b, (short)0, c, false, false);
}
__device__ __forceinline__ v8h cvt8(float4 a, float4 b) {
  v8h r;
  r[0]=(half_t)a.x; r[1]=(half_t)a.y; r[2]=(half_t)a.z; r[3]=(half_t)a.w;
  r[4]=(half_t)b.x; r[5]=(half_t)b.y; r[6]=(half_t)b.z; r[7]=(half_t)b.w;
  return r;
}

#if USE_ASYNC
// 16B per lane, global -> LDS, no VGPR data movement (ASYNCcnt-tracked)
__device__ __forceinline__ void async_cp16(const half_t* g, half_t* l) {
  __builtin_amdgcn_global_load_async_to_lds_b128(
      (__attribute__((address_space(1))) v4i_t*)(uintptr_t)g,
      (__attribute__((address_space(3))) v4i_t*)(uintptr_t)l,
      0, 0);
}
#endif

// ---- one-shot precision conversion: x fp32 -> f16 (same layout) ----
__launch_bounds__(256)
__global__ void cvt_f32_f16(const float* __restrict__ src, half_t* __restrict__ dst) {
  const size_t i = ((size_t)blockIdx.x * 256 + threadIdx.x) * 8;
  float4 f0 = *(const float4*)(src + i);
  float4 f1 = *(const float4*)(src + i + 4);
  *(v8h*)(dst + i) = cvt8(f0, f1);
}

// ---- one-shot weight conversion + transpose: W[k][n] fp32 -> Wt[n][k] f16 ----
__launch_bounds__(256)
__global__ void transpose_cvt(const float* __restrict__ W, half_t* __restrict__ Wt,
                              int Kdim, int Ncols) {
  __shared__ half_t tile[64][72];
  const int k0 = blockIdx.x * 64;
  const int n0 = blockIdx.y * 64;
  const int tid = threadIdx.x;
  {
    const int kr = tid >> 2;          // 0..63
    const int nc = (tid & 3) * 16;    // 0..48
    const float* p = W + (size_t)(k0 + kr) * Ncols + n0 + nc;
    float4 f0 = *(const float4*)(p + 0);
    float4 f1 = *(const float4*)(p + 4);
    float4 f2 = *(const float4*)(p + 8);
    float4 f3 = *(const float4*)(p + 12);
    *(v8h*)&tile[kr][nc]     = cvt8(f0, f1);
    *(v8h*)&tile[kr][nc + 8] = cvt8(f2, f3);
  }
  __syncthreads();
  {
    const int nr = tid >> 2;          // 0..63
    const int kc = (tid & 3) * 16;    // 0..48
    v8h o0, o1;
    #pragma unroll
    for (int j = 0; j < 8; ++j) o0[j] = tile[kc + j][nr];
    #pragma unroll
    for (int j = 0; j < 8; ++j) o1[j] = tile[kc + 8 + j][nr];
    half_t* q = Wt + (size_t)(n0 + nr) * Kdim + k0 + kc;
    *(v8h*)(q + 0) = o0;
    *(v8h*)(q + 8) = o1;
  }
}

// MODE: 0 = Q-proj (RoPE -> f16 q)
//       1 = K-proj (RoPE -> fp32 cache + f16 K[s][h])
//       2 = V-proj (fp32 cache + f16 V^T[h][s])
//       3 = O-proj (fp32 output)
// A: [M][Kdim] f16 row-major.  Bt: [Ncols][Kdim] f16 (pre-transposed).
template<int MODE>
__launch_bounds__(256, 2)
__global__ void gemm_fused(const half_t* __restrict__ Ah,
                           const half_t* __restrict__ Bt,
                           float* __restrict__ out32,
                           half_t* __restrict__ outh,
                           const int* __restrict__ positions,
                           const int* __restrict__ slots,
                           int Kdim, int Ncols)
{
  __shared__ union {
    struct { half_t a[2][128][40]; half_t bt[2][128][40]; } st;  // ping-pong f16 staging
    float c[128][132];                                           // fp32 epilogue tile
  } sh;

  const int tid   = threadIdx.x;       // 0..255 (8 waves)
  const int lane  = tid & 31;
  const int wid   = tid >> 5;
  const int wm    = wid & 1;           // 2 wave-rows of 64
  const int wn    = wid >> 1;          // 4 wave-cols of 32
  const int row0  = blockIdx.y * 128;
  const int col0  = blockIdx.x * 128;

  const int hl    = lane & 15;
  const int upper = lane >> 4;
  const int akoff = upper ? 8  : 0;    // A-fragment K base (halves)
  const int bkoff = upper ? 8  : 0;    // B-fragment K base (halves)

  // per-thread staging slice: 1 row, 32 bytes
  const int srow = tid >> 1;           // 0..127
  const int scol = (tid & 1) * 16;     // halves: 0 or 16
  const half_t* agp = Ah + (size_t)(row0 + srow) * Kdim + scol;
  const half_t* bgp = Bt + (size_t)(col0 + srow) * Kdim + scol;

  v8f acc[4][2] = {};
  int cur = 0;

#if USE_ASYNC
  // prologue: tile 0 into buffer 0 (4 async ops / wave)
  async_cp16(agp + 0, &sh.st.a [0][srow][scol]);
  async_cp16(agp + 8, &sh.st.a [0][srow][scol + 8]);
  async_cp16(bgp + 0, &sh.st.bt[0][srow][scol]);
  async_cp16(bgp + 8, &sh.st.bt[0][srow][scol + 8]);
#endif

  for (int k0 = 0; k0 < Kdim; k0 += 32) {
    const bool has_next = (k0 + 32 < Kdim);
#if USE_ASYNC
    if (has_next) {        // fetch next tile into other buffer, then wait current
      const half_t* an = agp + k0 + 32;
      const half_t* bn = bgp + k0 + 32;
      async_cp16(an + 0, &sh.st.a [cur ^ 1][srow][scol]);
      async_cp16(an + 8, &sh.st.a [cur ^ 1][srow][scol + 8]);
      async_cp16(bn + 0, &sh.st.bt[cur ^ 1][srow][scol]);
      async_cp16(bn + 8, &sh.st.bt[cur ^ 1][srow][scol + 8]);
      __builtin_amdgcn_s_wait_asynccnt(4);   // oldest 4 (current tile) done
    } else {
      __builtin_amdgcn_s_wait_asynccnt(0);
    }
#else
    {
      const half_t* an = agp + k0;
      const half_t* bn = bgp + k0;
      *(v8h*)&sh.st.a [cur][srow][scol]     = *(const v8h*)(an);
      *(v8h*)&sh.st.a [cur][srow][scol + 8] = *(const v8h*)(an + 8);
      *(v8h*)&sh.st.bt[cur][srow][scol]     = *(const v8h*)(bn);
      *(v8h*)&sh.st.bt[cur][srow][scol + 8] = *(const v8h*)(bn + 8);
      if (has_next) {
        __builtin_prefetch((const void*)(an + 32), 0, 1);
        __builtin_prefetch((const void*)(bn + 32), 0, 1);
      }
    }
#endif
    __syncthreads();

    v16h af[4];
    #pragma unroll
    for (int fm = 0; fm < 4; ++fm) {
      const half_t* p = &sh.st.a[cur][wm*64 + fm*16 + hl][0];
      af[fm] = cat16(*(const v8h*)(p + akoff), *(const v8h*)(p + akoff + 16));
    }
    v16h bf[2];
    #pragma unroll
    for (int fn = 0; fn < 2; ++fn) {
      const half_t* p = &sh.st.bt[cur][wn*32 + fn*16 + hl][bkoff];
      bf[fn] = cat16(*(const v8h*)p, *(const v8h*)(p + 16));
    }
    #pragma unroll
    for (int fm = 0; fm < 4; ++fm)
      #pragma unroll
      for (int fn = 0; fn < 2; ++fn)
        acc[fm][fn] = wmma_f16(af[fm], bf[fn], acc[fm][fn]);
    __syncthreads();
    cur ^= 1;
  }

  // ---- dump accumulators to fp32 LDS tile (C layout -> [row][col]) ----
  #pragma unroll
  for (int fm = 0; fm < 4; ++fm)
    #pragma unroll
    for (int fn = 0; fn < 2; ++fn)
      #pragma unroll
      for (int r = 0; r < 8; ++r)
        sh.c[wm*64 + fm*16 + upper*8 + r][wn*32 + fn*16 + hl] = acc[fm][fn][r];
  __syncthreads();

  // ---- fused epilogues ----
  if constexpr (MODE == 0 || MODE == 1) {
    const int head = blockIdx.x;              // 128-wide tile == one head
    for (int i = tid; i < 128 * 64; i += 256) {
      const int row = i >> 6;
      const int pr  = i & 63;                 // rotary pair index
      const int tg  = row0 + row;
      float x1 = sh.c[row][pr];
      float x2 = sh.c[row][pr + 64];
      float pos = (float)positions[tg];
      float inv = __expf((float)pr * (-2.0f / (float)HD) * LN_THETA);
      float ang = pos * inv;
      float cs = __cosf(ang), sn = __sinf(ang);
      float o1 = x1 * cs - x2 * sn;
      float o2 = x2 * cs + x1 * sn;
      if constexpr (MODE == 0) {
        half_t* q = outh + (size_t)tg * (NQ*HD) + head * HD;
        q[pr] = (half_t)o1; q[pr + 64] = (half_t)o2;
      } else {
        const int slot = slots[tg];
        float* ck = out32 + (size_t)slot * (NKV*HD) + head * HD;
        ck[pr] = o1; ck[pr + 64] = o2;
        const int b = slot >> 10, s = slot & (SEQ - 1);
        half_t* kf = outh + ((size_t)(b * NKV + head) * SEQ + s) * HD;
        kf[pr] = (half_t)o1; kf[pr + 64] = (half_t)o2;
      }
    }
  } else if constexpr (MODE == 2) {
    const int head = blockIdx.x;
    for (int i = tid; i < 128 * 128; i += 256) {
      const int row = i >> 7;
      const int h   = i & 127;
      const int tg  = row0 + row;
      const int slot = slots[tg];
      float val = sh.c[row][h];
      out32[(size_t)slot * (NKV*HD) + head * HD + h] = val;
      const int b = slot >> 10, s = slot & (SEQ - 1);
      outh[((size_t)(b * NKV + head) * HD + h) * SEQ + s] = (half_t)val;  // V^T
    }
  } else {
    for (int i = tid; i < (128 * 128) / 4; i += 256) {
      const int idx = i * 4;
      const int row = idx >> 7;
      const int c   = idx & 127;
      *(float4*)&out32[(size_t)(row0 + row) * Ncols + col0 + c] =
          *(const float4*)&sh.c[row][c];
    }
  }
}

// Flash attention: one wave32 per 16-query tile; kv-tiles of 32.
__launch_bounds__(128, 2)
__global__ void attn_fwd(const half_t* __restrict__ qh,
                         const half_t* __restrict__ kh,
                         const half_t* __restrict__ vt,
                         half_t* __restrict__ oh)
{
  __shared__ half_t pbuf[4][16][40];   // per-wave private P tile
  const int lane = threadIdx.x & 31;
  const int wid  = threadIdx.x >> 5;
  const int gw   = blockIdx.x * 4 + wid;
  const int qt  = gw & 63;             // q tile (S/16 = 64)
  const int n   = (gw >> 6) & 31;      // query head
  const int b   = gw >> 11;            // sequence
  const int kvh = n >> 2;              // GQA group of 4
  const int hl    = lane & 15;
  const int upper = lane >> 4;
  const int akoff = upper ? 8 : 0;
  const int bkoff = upper ? 8 : 0;

  const half_t* qbase = qh + (size_t)(b * SEQ + qt * 16 + hl) * (NQ*HD) + n * HD;
  v16h qf[4];
  #pragma unroll
  for (int kf = 0; kf < 4; ++kf)
    qf[kf] = cat16(*(const v8h*)(qbase + kf*32 + akoff),
                   *(const v8h*)(qbase + kf*32 + akoff + 16));

  v8f acc[8] = {};
  float mrun[8], lrun[8];
  #pragma unroll
  for (int r = 0; r < 8; ++r) { mrun[r] = -3.0e38f; lrun[r] = 0.0f; }

  const half_t* kbase = kh + (size_t)(b * NKV + kvh) * SEQ * HD;   // [s][h]
  const half_t* vbase = vt + (size_t)(b * NKV + kvh) * HD * SEQ;   // [h][s]

  const int qrow  = qt * 16 + upper * 8;          // + r -> query position
  const int ntile = (qt * 16 + 16 + 31) >> 5;     // causal tile bound

  for (int ti = 0; ti < ntile; ++ti) {
    const int s0 = ti * 32;
    v8f sc[2] = {};
    #pragma unroll
    for (int kf = 0; kf < 4; ++kf) {
      #pragma unroll
      for (int sg = 0; sg < 2; ++sg) {
        const half_t* kp = kbase + (size_t)(s0 + sg*16 + hl) * HD + kf*32 + bkoff;
        v16h bfrag = cat16(*(const v8h*)kp, *(const v8h*)(kp + 16));
        sc[sg] = wmma_f16(qf[kf], bfrag, sc[sg]);
      }
    }
    #pragma unroll
    for (int sg = 0; sg < 2; ++sg) {
      const int sp = s0 + sg * 16 + hl;
      #pragma unroll
      for (int r = 0; r < 8; ++r) {
        float v = sc[sg][r] * SM_SCALE;
        sc[sg][r] = (sp <= qrow + r) ? v : -3.0e38f;
      }
    }
    float mx[8];
    #pragma unroll
    for (int r = 0; r < 8; ++r) mx[r] = fmaxf(sc[0][r], sc[1][r]);
    #pragma unroll
    for (int d = 1; d < 16; d <<= 1)
      #pragma unroll
      for (int r = 0; r < 8; ++r)
        mx[r] = fmaxf(mx[r], __shfl_xor(mx[r], d, 32));
    float alpha[8];
    #pragma unroll
    for (int r = 0; r < 8; ++r) {
      float mn = fmaxf(mrun[r], mx[r]);
      alpha[r] = __expf(mrun[r] - mn);
      mrun[r] = mn;
    }
    float rs[8];
    #pragma unroll
    for (int r = 0; r < 8; ++r) {
      float p0 = __expf(sc[0][r] - mrun[r]);
      float p1 = __expf(sc[1][r] - mrun[r]);
      sc[0][r] = p0; sc[1][r] = p1;
      rs[r] = p0 + p1;
    }
    #pragma unroll
    for (int d = 1; d < 16; d <<= 1)
      #pragma unroll
      for (int r = 0; r < 8; ++r)
        rs[r] += __shfl_xor(rs[r], d, 32);
    #pragma unroll
    for (int r = 0; r < 8; ++r) lrun[r] = lrun[r] * alpha[r] + rs[r];
    #pragma unroll
    for (int hf = 0; hf < 8; ++hf)
      #pragma unroll
      for (int r = 0; r < 8; ++r)
        acc[hf][r] *= alpha[r];
    // P (C layout) -> LDS -> A fragment (per-wave DS is in-order, no barrier)
    #pragma unroll
    for (int sg = 0; sg < 2; ++sg)
      #pragma unroll
      for (int r = 0; r < 8; ++r)
        pbuf[wid][upper*8 + r][sg*16 + hl] = (half_t)sc[sg][r];
    const half_t* pp = &pbuf[wid][hl][0];
    v16h pf = cat16(*(const v8h*)(pp + akoff), *(const v8h*)(pp + akoff + 16));
    #pragma unroll
    for (int hf = 0; hf < 8; ++hf) {
      const half_t* vp = vbase + (size_t)(hf*16 + hl) * SEQ + s0 + bkoff;
      v16h vfrag = cat16(*(const v8h*)vp, *(const v8h*)(vp + 16));
      acc[hf] = wmma_f16(pf, vfrag, acc[hf]);
    }
  }

  #pragma unroll
  for (int r = 0; r < 8; ++r) lrun[r] = 1.0f / lrun[r];
  #pragma unroll
  for (int hf = 0; hf < 8; ++hf)
    #pragma unroll
    for (int r = 0; r < 8; ++r)
      oh[(size_t)(b * SEQ + qt * 16 + upper*8 + r) * (NQ*HD) + n * HD + hf*16 + hl]
          = (half_t)(acc[hf][r] * lrun[r]);
}

extern "C" void kernel_launch(void* const* d_in, const int* in_sizes, int n_in,
                              void* d_out, int out_size, void* d_ws, size_t ws_size,
                              hipStream_t stream) {
  (void)in_sizes; (void)n_in; (void)out_size; (void)ws_size;
  const float* x     = (const float*)d_in[1];
  const int*   pos   = (const int*)d_in[2];
  const int*   slots = (const int*)d_in[3];
  const float* Wq    = (const float*)d_in[4];
  const float* Wk    = (const float*)d_in[5];
  const float* Wv    = (const float*)d_in[6];
  const float* Wo    = (const float*)d_in[7];

  float* out     = (float*)d_out;
  float* cache_k = out;                                   // [T][K][H]
  float* cache_v = out + (size_t)TTOK * NKV * HD;
  float* o_out   = out + (size_t)2 * TTOK * NKV * HD;     // [T][D]

  char* ws = (char*)d_ws;
  half_t* x16   = (half_t*)ws;  ws += (size_t)TTOK * D_HID * 2;
  half_t* wq_t  = (half_t*)ws;  ws += (size_t)D_HID * (NQ*HD) * 2;
  half_t* wk_t  = (half_t*)ws;  ws += (size_t)D_HID * (NKV*HD) * 2;
  half_t* wv_t  = (half_t*)ws;  ws += (size_t)D_HID * (NKV*HD) * 2;
  half_t* wo_t  = (half_t*)ws;  ws += (size_t)(NQ*HD) * D_HID * 2;
  half_t* q16   = (half_t*)ws;  ws += (size_t)TTOK * D_HID * 2;
  half_t* k16   = (half_t*)ws;  ws += (size_t)BATCH * NKV * SEQ * HD * 2;
  half_t* vt16  = (half_t*)ws;  ws += (size_t)BATCH * NKV * SEQ * HD * 2;
  half_t* at16  = (half_t*)ws;

  dim3 blk(256);
  // one-shot precision conversion / weight transpose
  cvt_f32_f16<<<dim3(((size_t)TTOK * D_HID) / (256 * 8)), blk, 0, stream>>>(x, x16);
  transpose_cvt<<<dim3(D_HID/64, (NQ*HD)/64),  blk, 0, stream>>>(Wq, wq_t, D_HID, NQ*HD);
  transpose_cvt<<<dim3(D_HID/64, (NKV*HD)/64), blk, 0, stream>>>(Wk, wk_t, D_HID, NKV*HD);
  transpose_cvt<<<dim3(D_HID/64, (NKV*HD)/64), blk, 0, stream>>>(Wv, wv_t, D_HID, NKV*HD);
  transpose_cvt<<<dim3((NQ*HD)/64, D_HID/64),  blk, 0, stream>>>(Wo, wo_t, NQ*HD, D_HID);

  gemm_fused<0><<<dim3((NQ*HD)/128,  TTOK/128), blk, 0, stream>>>(
      x16, wq_t, nullptr, q16, pos, slots, D_HID, NQ*HD);
  gemm_fused<1><<<dim3((NKV*HD)/128, TTOK/128), blk, 0, stream>>>(
      x16, wk_t, cache_k, k16, pos, slots, D_HID, NKV*HD);
  gemm_fused<2><<<dim3((NKV*HD)/128, TTOK/128), blk, 0, stream>>>(
      x16, wv_t, cache_v, vt16, pos, slots, D_HID, NKV*HD);
  attn_fwd<<<dim3(BATCH * NQ * (SEQ/16) / 4), dim3(128), 0, stream>>>(
      q16, k16, vt16, at16);
  gemm_fused<3><<<dim3(D_HID/128, TTOK/128), blk, 0, stream>>>(
      at16, wo_t, o_out, nullptr, pos, slots, NQ*HD, D_HID);
}
